// Decoder_21973052686872
// MI455X (gfx1250) — matrix operand order repaired and checked by reading, not compiled, not run on hardware
//
#include <hip/hip_runtime.h>
#include <stdint.h>
#include <stddef.h>

typedef _Float16 half_t;
typedef __attribute__((ext_vector_type(16))) _Float16 v16h;
typedef __attribute__((ext_vector_type(8)))  float        v8f;
typedef __attribute__((ext_vector_type(4)))  unsigned int v4u;
typedef __attribute__((ext_vector_type(8)))  int          v8i;
typedef __attribute__((ext_vector_type(4)))  int          v4i;

#define IN_SZ 4096
#define SS    256     // state size
#define BB    64      // batch
#define TT    512     // time steps
#define HH    128     // hidden (S/2)

// ---------------------------------------------------------------------------
// WMMA helper: D = A(16x32 f16) * B(32x16 f16) + C(16x16 f32)
// ---------------------------------------------------------------------------
__device__ __forceinline__ v8f wmma16(v16h a, v16h b, v8f c) {
  return __builtin_amdgcn_wmma_f32_16x16x32_f16(
      /*neg_a=*/false, a, /*neg_b=*/false, b,
      /*c_mod=*/(short)0, c, /*reuse_a=*/false, /*reuse_b=*/false);
}

// A fragment (16x32 f16) from row-major [M][K] f16 row pointer.
// lane layout: halves[0:16) = K[kc*32 + h*16 .. +16)
__device__ __forceinline__ v16h load_a_f16(const half_t* __restrict__ row,
                                           int kc, int h) {
  const half_t* p0 = row + kc * 32 + h * 16;
  const half_t* p1 = p0 + 8;
  v16h a;
#pragma unroll
  for (int i = 0; i < 8; ++i) { a[i] = p0[i]; a[8 + i] = p1[i]; }
  return a;
}

// B fragment (32x16 f16) from row-major [N][K] f16 row pointer (col n = lane&15).
// lane layout: halves[0:16) = K[kc*32 + h*16 .. +16)  (one contiguous 32B run)
__device__ __forceinline__ v16h load_b_f16(const half_t* __restrict__ row,
                                           int kc, int h) {
  const half_t* p = row + kc * 32 + h * 16;
  v16h b;
#pragma unroll
  for (int i = 0; i < 16; ++i) b[i] = p[i];
  return b;
}

// Split-f16 loads from f32 memory (hi = f16(x), lo = f16(x - hi))
__device__ __forceinline__ void load_a_split(const float* __restrict__ row,
                                             int k0, int h, v16h& hi, v16h& lo) {
  const float* p0 = row + k0 + h * 16;
  const float* p1 = p0 + 8;
#pragma unroll
  for (int i = 0; i < 8; ++i) {
    float f = p0[i]; half_t x = (half_t)f;
    hi[i] = x; lo[i] = (half_t)(f - (float)x);
    float g = p1[i]; half_t y = (half_t)g;
    hi[8 + i] = y; lo[8 + i] = (half_t)(g - (float)y);
  }
}
__device__ __forceinline__ void load_b_split(const float* __restrict__ row,
                                             int k0, int h, v16h& hi, v16h& lo) {
  const float* p = row + k0 + h * 16;
#pragma unroll
  for (int i = 0; i < 16; ++i) {
    float f = p[i]; half_t x = (half_t)f;
    hi[i] = x; lo[i] = (half_t)(f - (float)x);
  }
}

// ---------------------------------------------------------------------------
// Kernel 1: ext[B*T, S] = x @ W_in^T + b_in   (bandwidth-bound, split-f16 WMMA)
// Workgroup tile: 64(M) x 256(N); wave tile: 32(M) x 64(N) -> 2x4 C frags.
// ---------------------------------------------------------------------------
__global__ __launch_bounds__(256, 2)
void ext_gemm(const float* __restrict__ x, const float* __restrict__ Win,
              const float* __restrict__ bin, float* __restrict__ ext) {
  const int tid  = threadIdx.x;
  const int wid  = tid >> 5;
  const int lane = tid & 31;
  const int hh   = lane >> 4;
  const int r    = lane & 15;
  const int mi   = wid & 1;        // 2 M strips of 32
  const int ni   = wid >> 1;       // 4 N strips of 64
  const int m0   = blockIdx.x * 64 + mi * 32;
  const int n0   = ni * 64;

  v8f acc[2][4];
#pragma unroll
  for (int a = 0; a < 2; ++a)
#pragma unroll
    for (int b = 0; b < 4; ++b)
#pragma unroll
      for (int v = 0; v < 8; ++v) acc[a][b][v] = 0.0f;

  for (int kc = 0; kc < IN_SZ / 32; ++kc) {
    const int k0 = kc * 32;
    v16h ahi[2], alo[2], bhi[4], blo[4];
#pragma unroll
    for (int mt = 0; mt < 2; ++mt)
      load_a_split(x + (size_t)(m0 + mt * 16 + r) * IN_SZ, k0, hh, ahi[mt], alo[mt]);
#pragma unroll
    for (int nt = 0; nt < 4; ++nt)
      load_b_split(Win + (size_t)(n0 + nt * 16 + r) * IN_SZ, k0, hh, bhi[nt], blo[nt]);
#pragma unroll
    for (int mt = 0; mt < 2; ++mt)
#pragma unroll
      for (int nt = 0; nt < 4; ++nt) {
        acc[mt][nt] = wmma16(ahi[mt], bhi[nt], acc[mt][nt]);
        acc[mt][nt] = wmma16(ahi[mt], blo[nt], acc[mt][nt]);
        acc[mt][nt] = wmma16(alo[mt], bhi[nt], acc[mt][nt]);
      }
  }

#pragma unroll
  for (int mt = 0; mt < 2; ++mt)
#pragma unroll
    for (int nt = 0; nt < 4; ++nt) {
      const int n = n0 + nt * 16 + r;
      const float bv = bin[n];
#pragma unroll
      for (int v = 0; v < 8; ++v) {
        const int m = m0 + mt * 16 + hh * 8 + v;
        ext[(size_t)m * SS + n] = acc[mt][nt][v] + bv;
      }
    }
}

// ---------------------------------------------------------------------------
// TDM: async-load a [16 rows x 256 f32] tile (row stride T*S elems) into LDS.
// D# built per CDNA5 ISA (2-D, data_size=4B).
// amdgpu-toolchain (clang-23) builtin form: 6 args, extra int32x8 zero-filled.
// ---------------------------------------------------------------------------
__device__ __forceinline__ void tdm_load_tile(const float* gptr, uint32_t ldsaddr) {
  const uint64_t ga = (uint64_t)(uintptr_t)gptr;
  v4u g0;
  g0[0] = 1u;                                   // count=1, user desc, no gather
  g0[1] = ldsaddr;                              // lds_addr
  g0[2] = (uint32_t)ga;                         // global_addr[31:0]
  g0[3] = (uint32_t)((ga >> 32) & 0x01FFFFFFu)  // global_addr[56:32]
        | 0x80000000u;                          // type = 2 ("image")
  v8i g1;
  g1[0] = (int)(2u << 16);                      // data_size = 4B
  g1[1] = (int)((unsigned)SS << 16);            // tensor_dim0 = 256 (bits 79:48)
  g1[2] = (int)(16u << 16);                     // tensor_dim1 = 16  (bits 111:80)
  g1[3] = (int)((unsigned)SS << 16);            // tile_dim0 = 256   (bits 127:112)
  g1[4] = 16;                                   // tile_dim1 = 16, tile_dim2 = 0
  g1[5] = (int)(TT * SS);                       // tensor_dim0_stride = T*S
  g1[6] = 0;
  g1[7] = 0;
  v4i g2 = {0, 0, 0, 0};                        // 2-D tensor: group 2 unused
  v4i g3 = {0, 0, 0, 0};                        // 2-D tensor: group 3 unused
  v8i g4 = {0, 0, 0, 0, 0, 0, 0, 0};            // clang-23 extra group (zeroed)
  __builtin_amdgcn_tensor_load_to_lds(g0, g1, g2, g3, g4, 0);
}

// ---------------------------------------------------------------------------
// Kernel 2: the recurrent scan. 4 WGs x 16 batch rows; all weights in LDS.
// Per step: A = f16(state + e_t); state = relu(A @ Wrec^T + brec);
//           h = relu(state @ Wout^T + bout); o = h @ Wout2^T + bout2.
// ---------------------------------------------------------------------------
__global__ __launch_bounds__(256, 1)
void rnn_scan(const float* __restrict__ ext,
              const float* __restrict__ Wrec, const float* __restrict__ brec,
              const float* __restrict__ Wout, const float* __restrict__ bout,
              const float* __restrict__ Wout2, const float* __restrict__ bout2,
              float* __restrict__ out) {
  __shared__ half_t sWrec[SS][SS];     // 128 KB  [n][k]
  __shared__ half_t sWout[HH][SS];     //  64 KB  [n][k]
  __shared__ half_t sA[16][SS];        //   8 KB  f16(state + e_t)
  __shared__ half_t sSt[16][SS];       //   8 KB  state (f16)
  __shared__ half_t sH[16][HH];        //   4 KB
  __shared__ float  sE[2][16 * SS];    //  32 KB  e_t double buffer (TDM target)
  __shared__ float  sBrec[SS];
  __shared__ float  sBout[HH];

  const int tid  = threadIdx.x;
  const int wid  = tid >> 5;
  const int lane = tid & 31;
  const int hh   = lane >> 4;
  const int r    = lane & 15;
  const int g    = blockIdx.x;          // 4 WGs, 16 batch rows each

  // ---- preload weights / biases, zero the state -------------------------
  for (int i = tid; i < SS * SS; i += 256) ((half_t*)sWrec)[i] = (half_t)Wrec[i];
  for (int i = tid; i < HH * SS; i += 256) ((half_t*)sWout)[i] = (half_t)Wout[i];
  for (int i = tid; i < SS; i += 256) sBrec[i] = brec[i];
  for (int i = tid; i < HH; i += 256) sBout[i] = bout[i];
  for (int i = tid; i < 16 * SS; i += 256) ((half_t*)sSt)[i] = (half_t)0.0f;

  const float* eg = ext + (size_t)g * 16 * TT * SS;  // this WG's batch slice

  // kick off TDM for t = 0
  if (wid == 0) tdm_load_tile(eg + 0 * SS, (uint32_t)(uintptr_t)&sE[0][0]);
  __syncthreads();

  for (int t = 0; t < TT; ++t) {
    const int buf = t & 1;

    // ---- prefetch e_{t+1}, then make sure e_t has landed ----------------
    if (wid == 0) {
      if (t + 1 < TT) {
        tdm_load_tile(eg + (size_t)(t + 1) * SS,
                      (uint32_t)(uintptr_t)&sE[(t + 1) & 1][0]);
        __builtin_amdgcn_s_wait_tensorcnt(1);  // in-order: load(t) done
      } else {
        __builtin_amdgcn_s_wait_tensorcnt(0);
      }
    }
    __syncthreads();

    // ---- phase 1: sA = f16(state + e_t) ---------------------------------
    for (int i = tid; i < 16 * SS; i += 256)
      ((half_t*)sA)[i] = (half_t)((float)((half_t*)sSt)[i] + sE[buf][i]);
    __syncthreads();

    // ---- phase 2: state = relu(sA @ Wrec^T + brec)  (M=16,N=256,K=256) --
    {
      v8f acc0, acc1;
#pragma unroll
      for (int v = 0; v < 8; ++v) { acc0[v] = 0.0f; acc1[v] = 0.0f; }
      const half_t* arow = &sA[r][0];
      const half_t* b0r  = &sWrec[wid * 32 + r][0];
      const half_t* b1r  = &sWrec[wid * 32 + 16 + r][0];
#pragma unroll
      for (int kc = 0; kc < 8; ++kc) {
        v16h a  = load_a_f16(arow, kc, hh);
        v16h b0 = load_b_f16(b0r, kc, hh);
        v16h b1 = load_b_f16(b1r, kc, hh);
        acc0 = wmma16(a, b0, acc0);
        acc1 = wmma16(a, b1, acc1);
      }
      const int c0 = wid * 32 + r;
      const int c1 = c0 + 16;
      const float bb0 = sBrec[c0], bb1 = sBrec[c1];
#pragma unroll
      for (int v = 0; v < 8; ++v) {
        const int m = hh * 8 + v;
        sSt[m][c0] = (half_t)fmaxf(acc0[v] + bb0, 0.0f);
        sSt[m][c1] = (half_t)fmaxf(acc1[v] + bb1, 0.0f);
      }
    }
    __syncthreads();

    // ---- phase 3: h = relu(state @ Wout^T + bout)  (M=16,N=128,K=256) ---
    {
      v8f acc;
#pragma unroll
      for (int v = 0; v < 8; ++v) acc[v] = 0.0f;
      const half_t* arow = &sSt[r][0];
      const half_t* brow = &sWout[wid * 16 + r][0];
#pragma unroll
      for (int kc = 0; kc < 8; ++kc) {
        v16h a = load_a_f16(arow, kc, hh);
        v16h b = load_b_f16(brow, kc, hh);
        acc = wmma16(a, b, acc);
      }
      const int col = wid * 16 + r;
      const float bb = sBout[col];
#pragma unroll
      for (int v = 0; v < 8; ++v)
        sH[hh * 8 + v][col] = (half_t)fmaxf(acc[v] + bb, 0.0f);
    }
    __syncthreads();

    // ---- phase 4: o = h @ Wout2^T + bout2  (N=2, scalar, wave 0) --------
    if (tid < 32) {
      const int bl = tid >> 1, j = tid & 1;
      float acco = bout2[j];
#pragma unroll 8
      for (int k = 0; k < HH; ++k)
        acco += (float)sH[bl][k] * Wout2[j * HH + k];
      out[((size_t)(g * 16 + bl) * TT + t) * 2 + j] = acco;
    }
    // next-iteration barriers provide the needed ordering
  }
}

// ---------------------------------------------------------------------------
extern "C" void kernel_launch(void* const* d_in, const int* in_sizes, int n_in,
                              void* d_out, int out_size, void* d_ws, size_t ws_size,
                              hipStream_t stream) {
  (void)in_sizes; (void)n_in; (void)out_size; (void)ws_size;
  const float* x     = (const float*)d_in[0];
  const float* Win   = (const float*)d_in[1];
  const float* bin   = (const float*)d_in[2];
  const float* Wrec  = (const float*)d_in[3];
  const float* brec  = (const float*)d_in[4];
  const float* Wout  = (const float*)d_in[5];
  const float* bout  = (const float*)d_in[6];
  const float* Wout2 = (const float*)d_in[7];
  const float* bout2 = (const float*)d_in[8];
  float* out = (float*)d_out;
  float* ext = (float*)d_ws;   // [B*T, S] f32 = 32 MB scratch

  ext_gemm<<<dim3((BB * TT) / 64), dim3(256), 0, stream>>>(x, Win, bin, ext);
  rnn_scan<<<dim3(4), dim3(256), 0, stream>>>(ext, Wrec, brec, Wout, bout,
                                              Wout2, bout2, out);
}
